// RoPEAttention_1022202216936
// MI455X (gfx1250) — matrix operand order repaired and checked by reading, not compiled, not run on hardware
//
#include <hip/hip_runtime.h>

// ---------------------------------------------------------------------------
// RoPE causal attention for MI455X (gfx1250, wave32, WMMA).
//
// Roofline: ~60 GFLOP total vs ~100MB HBM traffic (23.3 TB/s) -> compute
// bound on the matrix pipes. bf16 v_wmma_f32_16x16x32_bf16 (16K flops/inst)
// is ~8x the per-instruction throughput of the f32 16x16x4 WMMA, so convert
// fp32 inputs to bf16 for A/B and accumulate in fp32.
//
// GEMM blocking: 32x64 output per wave (2 A-frags x 4 B-frags -> 8 wmma per
// 32-wide K step) so each v_wmma costs ~3 b128 loads + ~6 cvt instead of ~8/16.
// ---------------------------------------------------------------------------

#define SEQ   2048
#define BSZ   2
#define HEADS 16
#define DK    64
#define DM    1024
#define NROW  (SEQ * BSZ)          // 4096 GEMM rows

typedef __attribute__((ext_vector_type(16))) __bf16 v16bf;
typedef __attribute__((ext_vector_type(8)))  float  v8f;

static __device__ inline v8f wmma_bf16(v16bf a, v16bf b, v8f c) {
  // D = A(16x32 bf16) * B(32x16 bf16) + C(16x16 f32)
  return __builtin_amdgcn_wmma_f32_16x16x32_bf16(
      /*neg_a=*/false, a, /*neg_b=*/false, b,
      /*c_mod=*/(short)0, c, /*reuse_a=*/false, /*reuse_b=*/false);
}

// A frag (16x32): lane<16 -> row=l16, K in {k0..k0+7, k0+16..k0+23}
//                 lane>=16 -> row=l16, K in {k0+8..15, k0+24..31}
// => per lane: two contiguous 8-float chunks at k0+half*8 and k0+half*8+16.
static __device__ inline v16bf load_a_frag_f32(const float* rowp) {
  v16bf a;
#pragma unroll
  for (int i = 0; i < 8; ++i) a[i]     = (__bf16)rowp[i];
#pragma unroll
  for (int i = 0; i < 8; ++i) a[8 + i] = (__bf16)rowp[16 + i];
  return a;
}

// ---------------------------------------------------------------------------
// Kernel 1: q/k/v = x @ W^T + bias.  One wave computes a 32x64 tile:
// 2 A-fragments (rows) x 4 B-fragments (cols) -> 8 C fragments, 8 wmma/K-step.
// B frag (32x16): col=l16, K in {k0+half*8..+7, k0+half*8+16..+23} (matches A).
// B column j is row j of W (we need W^T) -> two contiguous 8-float runs per lane.
// ---------------------------------------------------------------------------
__global__ __launch_bounds__(256) void qkv_gemm(
    const float* __restrict__ x,
    const float* __restrict__ Wq, const float* __restrict__ bq,
    const float* __restrict__ Wk, const float* __restrict__ bk,
    const float* __restrict__ Wv, const float* __restrict__ bv,
    float* __restrict__ qf, float* __restrict__ kf, float* __restrict__ vf)
{
  const int lane = threadIdx.x & 31;
  const int wv   = threadIdx.x >> 5;
  const int half = lane >> 4;
  const int l16  = lane & 15;

  const float* W; const float* bias; float* out;
  if (blockIdx.z == 0)      { W = Wq; bias = bq; out = qf; }
  else if (blockIdx.z == 1) { W = Wk; bias = bk; out = kf; }
  else                      { W = Wv; bias = bv; out = vf; }

  const int row0 = blockIdx.x * 32;
  const int col0 = (blockIdx.y * 8 + wv) * 64;

  v8f c[2][4];
#pragma unroll
  for (int rt = 0; rt < 2; ++rt)
#pragma unroll
    for (int ct = 0; ct < 4; ++ct) { v8f z = {}; c[rt][ct] = z; }

  const float* arow0 = x + (size_t)(row0 + l16)      * DM;
  const float* arow1 = x + (size_t)(row0 + 16 + l16) * DM;
  const float* brow[4];
#pragma unroll
  for (int ct = 0; ct < 4; ++ct)
    brow[ct] = W + (size_t)(col0 + ct * 16 + l16) * DM;

  for (int k0 = 0; k0 < DM; k0 += 32) {
    const float* ap0 = arow0 + k0 + half * 8;
    const float* ap1 = arow1 + k0 + half * 8;
    __builtin_prefetch(ap0 + 32, 0, 0);   // gfx1250 global_prefetch
    __builtin_prefetch(ap1 + 32, 0, 0);
    const v16bf a0 = load_a_frag_f32(ap0);
    const v16bf a1 = load_a_frag_f32(ap1);

#pragma unroll
    for (int ct = 0; ct < 4; ++ct) {
      const float* bp = brow[ct] + k0 + half * 8;
      v16bf b;
#pragma unroll
      for (int i = 0; i < 16; ++i) b[i] = (__bf16)bp[i < 8 ? i : i + 8];
      c[0][ct] = wmma_bf16(a0, b, c[0][ct]);
      c[1][ct] = wmma_bf16(a1, b, c[1][ct]);
    }
  }

#pragma unroll
  for (int ct = 0; ct < 4; ++ct) {
    const int col = col0 + ct * 16 + l16;
    const float bs = bias[col];
#pragma unroll
    for (int rt = 0; rt < 2; ++rt) {
#pragma unroll
      for (int vi = 0; vi < 8; ++vi) {
        const int row = row0 + rt * 16 + half * 8 + vi; // C: M=vi+8*half, N=l16
        out[(size_t)row * DM + col] = c[rt][ct][vi] + bs;
      }
    }
  }
}

// ---------------------------------------------------------------------------
// Kernel 2: RoPE (faithful theta = d / base^(2i), overflowing to 0 in fp32)
// + fp32 -> bf16 pack into attention-friendly layouts:
//   Qb[b][h][s][d], Kb[b][h][s][d], Vt[b][h][d][s]
// ---------------------------------------------------------------------------
__global__ __launch_bounds__(256) void rope_pack(
    const float* __restrict__ qf, const float* __restrict__ kf,
    const float* __restrict__ vf,
    __bf16* __restrict__ Qb, __bf16* __restrict__ Kb, __bf16* __restrict__ Vt)
{
  const int idx = blockIdx.x * blockDim.x + threadIdx.x;
  if (idx >= NROW * DM) return;
  const int c = idx % DM;
  const int r = idx / DM;          // r = s*BSZ + b  (x is [seq, bsz, dm])
  const int b = r % BSZ;
  const int s = r / BSZ;
  const int h  = c / DK;
  const int hd = c % DK;
  const int i2 = hd & 31;

  // theta = 1 / (base^(2*i2) / d); base^(2*i2) overflows fp32 -> inf -> theta 0
  const float p     = __powf(10000.0f, (float)(2 * i2));
  const float theta = 1.0f / (p / (float)DK);
  const float ang   = (float)s * theta;
  const float cs = cosf(ang), sn = sinf(ang);

  const int   pc  = (hd < 32) ? (c + 32) : (c - 32);
  const float sgn = (hd < 32) ? -1.0f : 1.0f;

  const float qv = qf[idx], qp = qf[(size_t)r * DM + pc];
  const float kv = kf[idx], kp = kf[(size_t)r * DM + pc];
  const float qr = qv * cs + sgn * qp * sn;
  const float kr = kv * cs + sgn * kp * sn;

  const size_t bh = (size_t)(b * HEADS + h);
  Qb[(bh * SEQ + s) * DK + hd] = (__bf16)qr;
  Kb[(bh * SEQ + s) * DK + hd] = (__bf16)kr;
  Vt[(bh * DK + hd) * SEQ + s] = (__bf16)vf[idx];
}

// ---------------------------------------------------------------------------
// Kernel 3: flash-style causal attention with softmax-one.
// One wave owns a 16-row tile of one (b,h). Per 32-column chunk:
//   S = Q K^T  (4 wmma), causal mask, streaming max/sum (shfl_xor within
//   16-lane halves matches the C-fragment row mapping), P transposed via
//   per-wave LDS slice into an A fragment, then acc += P V (4 wmma).
// Final: out = acc / (1 + l)  ==  e^{x-M} / (1 + sum e^{x-M})  (softmax-one).
// ---------------------------------------------------------------------------
__global__ __launch_bounds__(256) void attn(
    const __bf16* __restrict__ Qb, const __bf16* __restrict__ Kb,
    const __bf16* __restrict__ Vt, float* __restrict__ out)
{
  __shared__ __bf16 psh[8][16 * 32];   // per-wave P staging (transpose)

  const int lane = threadIdx.x & 31;
  const int wv   = threadIdx.x >> 5;
  const int half = lane >> 4;
  const int l16  = lane & 15;

  const int id      = blockIdx.x * 8 + wv;   // [0, 4096)
  const int rowTile = id & 127;
  const int bh      = id >> 7;               // [0, 32) = b*HEADS + h
  const int i0      = rowTile * 16;

  const __bf16* Q = Qb + (size_t)bh * SEQ * DK;
  const __bf16* K = Kb + (size_t)bh * SEQ * DK;
  const __bf16* V = Vt + (size_t)bh * DK * SEQ;

  float m[8], l[8];
  v8f acc[4];
#pragma unroll
  for (int vi = 0; vi < 8; ++vi) { m[vi] = -1e30f; l[vi] = 0.0f; }
#pragma unroll
  for (int d = 0; d < 4; ++d) { v8f z = {}; acc[d] = z; }

  // Q A-fragments for K-dim chunks k0=0 and k0=32 (reused every j chunk).
  v16bf qa[2];
#pragma unroll
  for (int kk = 0; kk < 2; ++kk) {
    const __bf16* qp = Q + (size_t)(i0 + l16) * DK + kk * 32 + half * 8;
#pragma unroll
    for (int i = 0; i < 8; ++i) qa[kk][i]     = qp[i];
#pragma unroll
    for (int i = 0; i < 8; ++i) qa[kk][8 + i] = qp[16 + i];
  }

  const int iLast = i0 + 15;
  for (int jc = 0; jc <= iLast; jc += 32) {
    // ---- scores: two 16x16 tiles (cols jc..jc+15, jc+16..jc+31) ----
    v8f c0 = {}, c1 = {};
#pragma unroll
    for (int kk = 0; kk < 2; ++kk) {
      v16bf b0, b1;
      const __bf16* kp0 = K + (size_t)(jc + l16)      * DK + kk * 32 + half * 8;
      const __bf16* kp1 = K + (size_t)(jc + 16 + l16) * DK + kk * 32 + half * 8;
#pragma unroll
      for (int t = 0; t < 16; ++t) { b0[t] = kp0[t < 8 ? t : t + 8]; b1[t] = kp1[t < 8 ? t : t + 8]; }
      c0 = wmma_bf16(qa[kk], b0, c0);
      c1 = wmma_bf16(qa[kk], b1, c1);
    }

    // ---- streaming softmax-one update ----
    float p0[8], p1[8];
#pragma unroll
    for (int vi = 0; vi < 8; ++vi) {
      const int rowi = i0 + half * 8 + vi;
      const int j0 = jc + l16, j1 = j0 + 16;
      const float s0 = (j0 <= rowi) ? c0[vi] * 0.125f : -1e30f;
      const float s1 = (j1 <= rowi) ? c1[vi] * 0.125f : -1e30f;

      float cm = fmaxf(s0, s1);              // row-max across the 16 lanes
      cm = fmaxf(cm, __shfl_xor(cm, 1, 32)); // xor<=8 stays within the half
      cm = fmaxf(cm, __shfl_xor(cm, 2, 32));
      cm = fmaxf(cm, __shfl_xor(cm, 4, 32));
      cm = fmaxf(cm, __shfl_xor(cm, 8, 32));

      const float mn    = fmaxf(m[vi], cm);
      const float alpha = __expf(m[vi] - mn);
      p0[vi] = __expf(s0 - mn);
      p1[vi] = __expf(s1 - mn);

      float rs = p0[vi] + p1[vi];
      rs += __shfl_xor(rs, 1, 32);
      rs += __shfl_xor(rs, 2, 32);
      rs += __shfl_xor(rs, 4, 32);
      rs += __shfl_xor(rs, 8, 32);

      l[vi] = l[vi] * alpha + rs;
      m[vi] = mn;
#pragma unroll
      for (int d = 0; d < 4; ++d) acc[d][vi] *= alpha;
    }

    // ---- transpose P (C layout -> A fragment) through this wave's LDS ----
    __bf16* pl = &psh[wv][0];
#pragma unroll
    for (int vi = 0; vi < 8; ++vi) {
      const int prow = half * 8 + vi;
      pl[prow * 32 + l16]      = (__bf16)p0[vi];
      pl[prow * 32 + 16 + l16] = (__bf16)p1[vi];
    }
    asm volatile("" ::: "memory");   // same-wave DS ops are in order on HW
    v16bf pa;
    {
      const __bf16* pp = pl + l16 * 32 + half * 8;
#pragma unroll
      for (int i = 0; i < 8; ++i) pa[i]     = pp[i];
#pragma unroll
      for (int i = 0; i < 8; ++i) pa[8 + i] = pp[16 + i];
    }
    asm volatile("" ::: "memory");

    // ---- acc += P @ V  (V^T rows are contiguous in j) ----
#pragma unroll
    for (int dt = 0; dt < 4; ++dt) {
      v16bf vb;
      const __bf16* vp = V + (size_t)(dt * 16 + l16) * SEQ + jc + half * 8;
#pragma unroll
      for (int t = 0; t < 16; ++t) vb[t] = vp[t < 8 ? t : t + 8];
      acc[dt] = wmma_bf16(pa, vb, acc[dt]);
    }
  }

  // ---- epilogue: softmax-one normalization, write [s, b, dm] fp32 ----
  const int b = bh >> 4;
  const int h = bh & 15;
#pragma unroll
  for (int vi = 0; vi < 8; ++vi) {
    const int rowi = i0 + half * 8 + vi;
    const float inv = 1.0f / (1.0f + l[vi]);
#pragma unroll
    for (int dt = 0; dt < 4; ++dt) {
      out[((size_t)rowi * BSZ + b) * DM + h * DK + dt * 16 + l16] =
          acc[dt][vi] * inv;
    }
  }
}

// ---------------------------------------------------------------------------
extern "C" void kernel_launch(void* const* d_in, const int* in_sizes, int n_in,
                              void* d_out, int out_size, void* d_ws, size_t ws_size,
                              hipStream_t stream) {
  (void)in_sizes; (void)n_in; (void)out_size; (void)ws_size;

  const float* x  = (const float*)d_in[0];
  const float* Wq = (const float*)d_in[1];
  const float* bq = (const float*)d_in[2];
  const float* Wk = (const float*)d_in[3];
  const float* bk = (const float*)d_in[4];
  const float* Wv = (const float*)d_in[5];
  const float* bv = (const float*)d_in[6];
  float* out = (float*)d_out;

  // Workspace layout (72 MB): fp32 q/k/v pre-RoPE, then bf16 packed Q/K/V^T.
  const size_t NE = (size_t)NROW * DM;            // 4M elements
  float*  qf = (float*)d_ws;
  float*  kf = qf + NE;
  float*  vf = kf + NE;
  __bf16* Qb = (__bf16*)(vf + NE);
  __bf16* Kb = Qb + NE;
  __bf16* Vt = Kb + NE;

  // 1) QKV projections: 128 row tiles (32 rows) x 16 col groups (64 cols) x 3
  dim3 ggrid(NROW / 32, (DM / 64) / 8, 3);
  qkv_gemm<<<ggrid, 256, 0, stream>>>(x, Wq, bq, Wk, bk, Wv, bv, qf, kf, vf);

  // 2) RoPE + bf16 pack
  const int nelem = NROW * DM;
  rope_pack<<<(nelem + 255) / 256, 256, 0, stream>>>(qf, kf, vf, Qb, Kb, Vt);

  // 3) Flash attention with softmax-one: 4096 waves = 512 blocks of 8 waves
  attn<<<(BSZ * HEADS * (SEQ / 16)) / 8, 256, 0, stream>>>(Qb, Kb, Vt, out);
}
